// ComplexViT_47356309406063
// MI455X (gfx1250) — compile-verified, matrix operand order checked by reading, not run on hardware
//
#include <hip/hip_runtime.h>
#include <hip/hip_bf16.h>
#include <math.h>

// ---------------- problem constants ----------------
#define B_      16
#define IMG_    224
#define P_      16
#define D_      768
#define H_      12
#define DEPTH_  12
#define MLP_    3072
#define NC_     1000
#define HD_     64
#define NTOK_   197      // 14*14 + 1
#define NTP_    224      // padded token count: 14 tiles of 16, K-chunks of 32
#define NPATCH_ 196
#define GRID14_ 14
#define PD_     256      // P*P*C

typedef __attribute__((ext_vector_type(16))) __bf16 bf16x16;
typedef __attribute__((ext_vector_type(8)))  __bf16 bf16x8;
typedef __attribute__((ext_vector_type(8)))  float  f32x8;

// ---------------- WMMA helpers ----------------
// 16-bit A/B fragment, K contiguous: lane L covers row (L&15), K-base kb0+((L>>4)<<3);
// elements [kb..kb+7] -> VGPR0-3, [kb+16..kb+23] -> VGPR4-7. Two 16-byte loads, no guards.
__device__ __forceinline__ bf16x16 frag_ld(const __bf16* p) {
  bf16x8 lo = *(const bf16x8*)(p);
  bf16x8 hi = *(const bf16x8*)(p + 16);
  return __builtin_shufflevector(lo, hi, 0, 1, 2, 3, 4, 5, 6, 7,
                                 8, 9, 10, 11, 12, 13, 14, 15);
}

__device__ __forceinline__ f32x8 wmma_bf16(bf16x16 a, bf16x16 b, f32x8 c) {
  return __builtin_amdgcn_wmma_f32_16x16x32_bf16(false, a, false, b, (short)0, c, false, false);
}

// ---------------- generic batched complex GEMM (bf16 in, f32 acc, bf16 out) ----------------
// C[m,n] = sum_k A[m,k] * B[n,k]  (complex; conj(B) if conj!=0; A real if !CPLXA)
// Each wave produces a 16x32 tile (two N-tiles sharing the A fragments).
// All dims tile-exact by construction: M%16==0, Nc%32==0, K%32==0. No bounds checks.
template <bool CPLXA>
__global__ void cgemm_bf16_kernel(
    const __bf16* __restrict__ Ar, const __bf16* __restrict__ Ai, int a_rs, size_t a_bs,
    const __bf16* __restrict__ Br, const __bf16* __restrict__ Bi, int b_rs, size_t b_bs,
    const float* __restrict__ bias,
    __bf16* __restrict__ Cr, __bf16* __restrict__ Ci, int c_rs, size_t c_bs,
    int M, int Nc, int K, int conj, int remap_n) {
  int tiles_n = Nc >> 5;                            // 32-wide column tiles
  int tiles_m = M >> 4;
  int wave = blockIdx.x * (blockDim.x >> 5) + (threadIdx.x >> 5);
  if (wave >= tiles_m * tiles_n) return;            // wave-uniform exit
  int tm = wave / tiles_n;
  int tn = wave - tm * tiles_n;

  size_t bofs = (size_t)blockIdx.y;
  Ar += bofs * a_bs;
  Br += bofs * b_bs;  Bi += bofs * b_bs;
  Cr += bofs * c_bs;  Ci += bofs * c_bs;

  int lane  = threadIdx.x & 31;
  int l15   = lane & 15;
  int kb_hi = (lane >> 4) << 3;
  int r0 = tm << 4, c0 = tn << 5;

  const __bf16* pAr = Ar + (size_t)(r0 + l15) * a_rs + kb_hi;
  const __bf16* pAi = pAr;
  if (CPLXA) pAi = Ai + bofs * a_bs + (size_t)(r0 + l15) * a_rs + kb_hi;
  const __bf16* pBr0 = Br + (size_t)(c0 + l15) * b_rs + kb_hi;
  const __bf16* pBi0 = Bi + (size_t)(c0 + l15) * b_rs + kb_hi;
  const __bf16* pBr1 = pBr0 + (size_t)16 * b_rs;
  const __bf16* pBi1 = pBi0 + (size_t)16 * b_rs;

  f32x8 p1a{}, p2a{}, p3a{}, p4a{};
  f32x8 p1b{}, p2b{}, p3b{}, p4b{};
  for (int k0 = 0; k0 < K; k0 += 32) {
    __builtin_prefetch(pAr + 32);                   // speculative; safe past end
    __builtin_prefetch(pBr0 + 32);
    __builtin_prefetch(pBr1 + 32);
    bf16x16 fa_r  = frag_ld(pAr);
    bf16x16 fb_r0 = frag_ld(pBr0);
    bf16x16 fb_i0 = frag_ld(pBi0);
    bf16x16 fb_r1 = frag_ld(pBr1);
    bf16x16 fb_i1 = frag_ld(pBi1);
    p1a = wmma_bf16(fa_r, fb_r0, p1a);
    p1b = wmma_bf16(fa_r, fb_r1, p1b);
    p3a = wmma_bf16(fa_r, fb_i0, p3a);
    p3b = wmma_bf16(fa_r, fb_i1, p3b);
    if constexpr (CPLXA) {
      bf16x16 fa_i = frag_ld(pAi);
      p2a = wmma_bf16(fa_i, fb_i0, p2a);
      p2b = wmma_bf16(fa_i, fb_i1, p2b);
      p4a = wmma_bf16(fa_i, fb_r0, p4a);
      p4b = wmma_bf16(fa_i, fb_r1, p4b);
      pAi += 32;
    }
    pAr += 32; pBr0 += 32; pBi0 += 32; pBr1 += 32; pBi1 += 32;
  }

  // epilogue: Cr = p1 - s*p2, Ci = s*p3 + p4   (s = conj ? -1 : +1)
  float s = conj ? -1.0f : 1.0f;
#pragma unroll
  for (int half = 0; half < 2; ++half) {
    int col = c0 + (half << 4) + l15;
    f32x8 q1 = half ? p1b : p1a;
    f32x8 q2 = half ? p2b : p2a;
    f32x8 q3 = half ? p3b : p3a;
    f32x8 q4 = half ? p4b : p4a;
    float b_r = 0.0f, b_i = 0.0f;
    if (bias) { b_r = bias[2 * col]; b_i = bias[2 * col + 1]; }
#pragma unroll
    for (int r = 0; r < 8; ++r) {
      int m = r0 + r + kb_hi;                       // lanes 16-31 hold rows +8
      int orow = m;
      if (remap_n > 0) orow = (m / remap_n) * (remap_n + 1) + 1 + (m % remap_n);
      size_t idx = (size_t)orow * c_rs + col;
      Cr[idx] = (__bf16)(q1[r] - s * q2[r] + b_r);
      Ci[idx] = (__bf16)(s * q3[r] + q4[r] + b_i);
    }
  }
}

// ---------------- helper kernels ----------------
// deinterleave complex64 -> bf16 planes
__global__ void deint_kernel(const float* __restrict__ w,
                             __bf16* __restrict__ wr, __bf16* __restrict__ wi, int n) {
  int id = blockIdx.x * blockDim.x + threadIdx.x;
  if (id >= n) return;
  wr[id] = (__bf16)w[2 * id];
  wi[id] = (__bf16)w[2 * id + 1];
}

__global__ void im2col_kernel(const float* __restrict__ img, __bf16* __restrict__ patches) {
  int id = blockIdx.x * blockDim.x + threadIdx.x;
  int total = B_ * NPATCH_ * PD_;
  if (id >= total) return;
  int k = id % PD_;
  int p = (id / PD_) % NPATCH_;
  int b = id / (PD_ * NPATCH_);
  int py = k / P_, px = k % P_;
  int pi = p / GRID14_, pj = p % GRID14_;
  patches[id] = (__bf16)img[((size_t)b * IMG_ + pi * P_ + py) * IMG_ + pj * P_ + px];
}

__global__ void cls_kernel(const float* __restrict__ ct,
                           __bf16* __restrict__ xr, __bf16* __restrict__ xi) {
  int id = blockIdx.x * blockDim.x + threadIdx.x;
  if (id >= B_ * D_) return;
  int b = id / D_, d = id % D_;
  size_t o = (size_t)(b * NTOK_) * D_ + d;
  xr[o] = (__bf16)ct[2 * d];
  xi[o] = (__bf16)ct[2 * d + 1];
}

// qkv (B*N, 3D) -> q,k head-major (B,H,224,HD) with RoPE, v transposed (B,H,HD,224).
// Pad rows/cols written as zeros (deterministic, keeps GEMMs guard-free).
__global__ void qkv_rope_kernel(const __bf16* __restrict__ qkvr, const __bf16* __restrict__ qkvi,
                                const float* __restrict__ freqs,
                                __bf16* __restrict__ qr, __bf16* __restrict__ qi,
                                __bf16* __restrict__ kr, __bf16* __restrict__ ki,
                                __bf16* __restrict__ vtr, __bf16* __restrict__ vti) {
  int id = blockIdx.x * blockDim.x + threadIdx.x;
  int total = B_ * H_ * NTP_ * HD_;
  if (id >= total) return;
  int hd = id % HD_;
  int n  = (id / HD_) % NTP_;
  int h  = (id / (HD_ * NTP_)) % H_;
  int b  = id / (HD_ * NTP_ * H_);
  size_t qo = ((size_t)(b * H_ + h) * NTP_ + n) * HD_ + hd;   // (bh, n, hd)
  size_t vo = ((size_t)(b * H_ + h) * HD_ + hd) * NTP_ + n;   // (bh, hd, n) transposed
  if (n < NTOK_) {
    size_t row = (size_t)(b * NTOK_ + n) * (3 * D_);
    int cb = h * (HD_ * 3) + hd * 3;                // (3D) viewed as (H, HD, 3)
    float q_r = (float)qkvr[row + cb + 0], q_i = (float)qkvi[row + cb + 0];
    float k_r = (float)qkvr[row + cb + 1], k_i = (float)qkvi[row + cb + 1];
    float v_r = (float)qkvr[row + cb + 2], v_i = (float)qkvi[row + cb + 2];
    float fr = freqs[(n * HD_ + hd) * 2], fi = freqs[(n * HD_ + hd) * 2 + 1];
    qr[qo] = (__bf16)(q_r * fr - q_i * fi);  qi[qo] = (__bf16)(q_r * fi + q_i * fr);
    kr[qo] = (__bf16)(k_r * fr - k_i * fi);  ki[qo] = (__bf16)(k_r * fi + k_i * fr);
    vtr[vo] = (__bf16)v_r;                   vti[vo] = (__bf16)v_i;
  } else {
    qr[qo] = (__bf16)0.0f;  qi[qo] = (__bf16)0.0f;
    kr[qo] = (__bf16)0.0f;  ki[qo] = (__bf16)0.0f;
    vtr[vo] = (__bf16)0.0f; vti[vo] = (__bf16)0.0f;
  }
}

// softmax over |dots|*scale along last dim (197 valid of 224); attn (real) -> dr in place
__global__ void softmax_abs_kernel(__bf16* __restrict__ dr, const __bf16* __restrict__ di) {
  __shared__ float red[256];
  int row = blockIdx.x;                             // B*H*224 rows
  __bf16*       pr = dr + (size_t)row * NTP_;
  const __bf16* pi = di + (size_t)row * NTP_;
  int t = threadIdx.x;
  float v = -3.0e38f;
  if (t < NTOK_) {
    float a_r = (float)pr[t], a_i = (float)pi[t];
    v = sqrtf(a_r * a_r + a_i * a_i) * 0.125f;      // scale = HD^-0.5
  }
  red[t] = v; __syncthreads();
  for (int s = 128; s > 0; s >>= 1) { if (t < s) red[t] = fmaxf(red[t], red[t + s]); __syncthreads(); }
  float mx = red[0]; __syncthreads();
  float e = (t < NTOK_) ? __expf(v - mx) : 0.0f;
  red[t] = e; __syncthreads();
  for (int s = 128; s > 0; s >>= 1) { if (t < s) red[t] += red[t + s]; __syncthreads(); }
  float sum = red[0];
  if (t < NTOK_) pr[t] = (__bf16)(e / sum);
  else if (t < NTP_) pr[t] = (__bf16)0.0f;          // pad cols contribute zero to attn@v
}

// (B,H,224,HD) head-major -> (B*197, D) token-major
__global__ void merge_heads_kernel(const __bf16* __restrict__ ohr, const __bf16* __restrict__ ohi,
                                   __bf16* __restrict__ omr, __bf16* __restrict__ omi) {
  int id = blockIdx.x * blockDim.x + threadIdx.x;
  int total = B_ * H_ * NTOK_ * HD_;
  if (id >= total) return;
  int hd = id % HD_;
  int n  = (id / HD_) % NTOK_;
  int h  = (id / (HD_ * NTOK_)) % H_;
  int b  = id / (HD_ * NTOK_ * H_);
  size_t src = ((size_t)(b * H_ + h) * NTP_ + n) * HD_ + hd;
  size_t dst = (size_t)(b * NTOK_ + n) * D_ + h * HD_ + hd;
  omr[dst] = ohr[src];
  omi[dst] = ohi[src];
}

// feat = [Re(cls), Im(cls)] @ head_w^T + head_b -> (B, NC) fp32
__global__ void head_kernel(const __bf16* __restrict__ xr, const __bf16* __restrict__ xi,
                            const float* __restrict__ hw, const float* __restrict__ hb,
                            float* __restrict__ out) {
  int id = blockIdx.x * blockDim.x + threadIdx.x;
  if (id >= B_ * NC_) return;
  int b = id / NC_, o = id % NC_;
  const __bf16* pr = xr + (size_t)(b * NTOK_) * D_;
  const __bf16* pi = xi + (size_t)(b * NTOK_) * D_;
  const float*  w  = hw + (size_t)o * (2 * D_);
  float acc = hb[o];
  for (int d = 0; d < D_; ++d) acc += (float)pr[d] * w[d] + (float)pi[d] * w[D_ + d];
  out[id] = acc;
}

// ---------------- host-side orchestration ----------------
static inline void launch_cgemm(hipStream_t stream,
    const __bf16* Ar, const __bf16* Ai, int a_rs, size_t a_bs,
    const __bf16* Br, const __bf16* Bi, int b_rs, size_t b_bs,
    const float* bias,
    __bf16* Cr, __bf16* Ci, int c_rs, size_t c_bs,
    int M, int Nc, int K, int conj, int remap_n, int batches) {
  int tiles = (M / 16) * (Nc / 32);
  dim3 grid((tiles + 7) / 8, batches);
  if (Ai)
    cgemm_bf16_kernel<true><<<grid, 256, 0, stream>>>(Ar, Ai, a_rs, a_bs, Br, Bi, b_rs, b_bs,
                                                      bias, Cr, Ci, c_rs, c_bs,
                                                      M, Nc, K, conj, remap_n);
  else
    cgemm_bf16_kernel<false><<<grid, 256, 0, stream>>>(Ar, Ar, a_rs, a_bs, Br, Bi, b_rs, b_bs,
                                                       bias, Cr, Ci, c_rs, c_bs,
                                                       M, Nc, K, conj, remap_n);
}

static inline void launch_deint(hipStream_t stream, const float* w,
                                __bf16* wr, __bf16* wi, int n) {
  deint_kernel<<<(n + 255) / 256, 256, 0, stream>>>(w, wr, wi, n);
}

extern "C" void kernel_launch(void* const* d_in, const int* in_sizes, int n_in,
                              void* d_out, int out_size, void* d_ws, size_t ws_size,
                              hipStream_t stream) {
  (void)in_sizes; (void)n_in; (void)out_size; (void)ws_size;
  const float* img     = (const float*)d_in[0];
  const float* patch_w = (const float*)d_in[1];   // complex64 interleaved
  const float* patch_b = (const float*)d_in[2];
  const float* cls_t   = (const float*)d_in[3];
  const float* qkv_w   = (const float*)d_in[4];
  const float* qkv_b   = (const float*)d_in[5];
  const float* out_w   = (const float*)d_in[6];
  const float* out_b   = (const float*)d_in[7];
  const float* ff1_w   = (const float*)d_in[8];
  const float* ff1_b   = (const float*)d_in[9];
  const float* ff2_w   = (const float*)d_in[10];
  const float* ff2_b   = (const float*)d_in[11];
  const float* head_w  = (const float*)d_in[12];
  const float* head_b  = (const float*)d_in[13];
  const float* freqs   = (const float*)d_in[14];
  float* out = (float*)d_out;

  const int M = B_ * NTOK_;                        // 3152 = 197*16, tile-exact
  const size_t XSZ = (size_t)M * D_;               // 2,420,736
  const size_t ASZ = (size_t)M * MLP_;             // 9,682,944
  const size_t HP  = (size_t)B_ * H_ * NTP_ * HD_; // 2,752,512 (q/k/o planes, v^T planes)
  const size_t DP  = (size_t)B_ * H_ * NTP_ * NTP_;// 9,633,792

  __bf16* ws  = (__bf16*)d_ws;
  __bf16* xr  = ws;            __bf16* xi  = xr + XSZ;
  __bf16* bAr = xi + XSZ;      __bf16* bAi = bAr + ASZ;  // qkv out / mlp hidden
  __bf16* qhr = bAi + ASZ;     __bf16* qhi = qhr + HP;
  __bf16* khr = qhi + HP;      __bf16* khi = khr + HP;
  __bf16* vtr = khi + HP;      __bf16* vti = vtr + HP;   // v transposed (bh, hd, n)
  __bf16* ohr = vti + HP;      __bf16* ohi = ohr + HP;
  __bf16* ddr = ohi + HP;      __bf16* ddi = ddr + DP;   // dots; ddr becomes attn
  __bf16* omr = ddi + DP;      __bf16* omi = omr + XSZ;  // merged heads
  __bf16* pat = omi + XSZ;                               // 802,816
  __bf16* pwr = pat + (size_t)B_ * NPATCH_ * PD_;  __bf16* pwi = pwr + (size_t)D_ * PD_;
  __bf16* wqr = pwi + (size_t)D_ * PD_;            __bf16* wqi = wqr + (size_t)3 * D_ * D_;
  __bf16* wor = wqi + (size_t)3 * D_ * D_;         __bf16* woi = wor + (size_t)D_ * D_;
  __bf16* w1r = woi + (size_t)D_ * D_;             __bf16* w1i = w1r + (size_t)MLP_ * D_;
  __bf16* w2r = w1i + (size_t)MLP_ * D_;           __bf16* w2i = w2r + (size_t)D_ * MLP_;

  // 1) patch extraction + cls token + patch weight staging
  {
    int total = B_ * NPATCH_ * PD_;
    im2col_kernel<<<(total + 255) / 256, 256, 0, stream>>>(img, pat);
    cls_kernel<<<(B_ * D_ + 255) / 256, 256, 0, stream>>>(cls_t, xr, xi);
    launch_deint(stream, patch_w, pwr, pwi, D_ * PD_);
  }
  // 2) patch embed: x[:,1:] = patches @ patch_w^T + patch_b  (real A x complex B)
  launch_cgemm(stream, pat, nullptr, PD_, 0,
               pwr, pwi, PD_, 0,
               patch_b, xr, xi, D_, 0,
               B_ * NPATCH_, D_, PD_, 0, NPATCH_, 1);

  // 3) transformer layers
  for (int l = 0; l < DEPTH_; ++l) {
    // stage this layer's weights as bf16 planes (contiguous K)
    launch_deint(stream, qkv_w + (size_t)l * 3 * D_ * D_ * 2, wqr, wqi, 3 * D_ * D_);
    launch_deint(stream, out_w + (size_t)l * D_ * D_ * 2,     wor, woi, D_ * D_);
    launch_deint(stream, ff1_w + (size_t)l * MLP_ * D_ * 2,   w1r, w1i, MLP_ * D_);
    launch_deint(stream, ff2_w + (size_t)l * D_ * MLP_ * 2,   w2r, w2i, D_ * MLP_);
    const float* qb  = qkv_b + (size_t)l * 3 * D_ * 2;
    const float* ob  = out_b + (size_t)l * D_ * 2;
    const float* f1b = ff1_b + (size_t)l * MLP_ * 2;
    const float* f2b = ff2_b + (size_t)l * D_ * 2;

    // qkv = x @ qw^T + qb : (3152, 2304)
    launch_cgemm(stream, xr, xi, D_, 0,
                 wqr, wqi, D_, 0,
                 qb, bAr, bAi, 3 * D_, 0,
                 M, 3 * D_, D_, 0, 0, 1);

    // split heads + RoPE (+ zero padding, + v transpose)
    {
      int total = B_ * H_ * NTP_ * HD_;
      qkv_rope_kernel<<<(total + 255) / 256, 256, 0, stream>>>(
          bAr, bAi, freqs, qhr, qhi, khr, khi, vtr, vti);
    }

    // dots = q @ conj(k)^T per (b,h): batched (192), 224x224, K=64
    launch_cgemm(stream, qhr, qhi, HD_, (size_t)NTP_ * HD_,
                 khr, khi, HD_, (size_t)NTP_ * HD_,
                 nullptr, ddr, ddi, NTP_, (size_t)NTP_ * NTP_,
                 NTP_, NTP_, HD_, 1, 0, B_ * H_);

    // attn = softmax(|dots| * scale), in place into ddr (zeros in pad cols)
    softmax_abs_kernel<<<B_ * H_ * NTP_, 256, 0, stream>>>(ddr, ddi);

    // o = attn @ v : real A x complex B; v^T layout makes B K-contiguous
    launch_cgemm(stream, ddr, nullptr, NTP_, (size_t)NTP_ * NTP_,
                 vtr, vti, NTP_, (size_t)HD_ * NTP_,
                 nullptr, ohr, ohi, HD_, (size_t)NTP_ * HD_,
                 NTP_, HD_, NTP_, 0, 0, B_ * H_);

    // merge heads -> (3152, 768)
    {
      int total = B_ * H_ * NTOK_ * HD_;
      merge_heads_kernel<<<(total + 255) / 256, 256, 0, stream>>>(ohr, ohi, omr, omi);
    }

    // out projection -> x (x was consumed by qkv GEMM)
    launch_cgemm(stream, omr, omi, D_, 0,
                 wor, woi, D_, 0,
                 ob, xr, xi, D_, 0,
                 M, D_, D_, 0, 0, 1);

    // ff1: (3152, 3072)
    launch_cgemm(stream, xr, xi, D_, 0,
                 w1r, w1i, D_, 0,
                 f1b, bAr, bAi, MLP_, 0,
                 M, MLP_, D_, 0, 0, 1);

    // ff2: back to (3152, 768) -> new x
    launch_cgemm(stream, bAr, bAi, MLP_, 0,
                 w2r, w2i, MLP_, 0,
                 f2b, xr, xi, D_, 0,
                 M, D_, MLP_, 0, 0, 1);
  }

  // 4) classifier head on cls tokens
  head_kernel<<<(B_ * NC_ + 255) / 256, 256, 0, stream>>>(xr, xi, head_w, head_b, out);
}
